// CrossAttention_M_67808943669434
// MI455X (gfx1250) — matrix operand order repaired and checked by reading, not compiled, not run on hardware
//
#include <hip/hip_runtime.h>

#define BATCH 8
#define CH    192
#define CH3   576
#define HWSZ  16384
#define WIDTH 128
#define HEADS 8
#define HDIM  24

typedef __attribute__((ext_vector_type(16))) __bf16 v16bf;
typedef __attribute__((ext_vector_type(8)))  __bf16 v8bf;
typedef __attribute__((ext_vector_type(8)))  float  v8f;
typedef __attribute__((ext_vector_type(4)))  unsigned int u32x4;
typedef __attribute__((ext_vector_type(8)))  unsigned int u32x8;

union FragAB { v16bf v; v8bf h[2]; __bf16 e[16]; };
union FragC  { v8f v; float f[8]; };
union PkBf   { __bf16 h[2]; unsigned u; };

__device__ __forceinline__ __bf16 f2bf(float f) { return (__bf16)f; }
__device__ __forceinline__ float  bf2f(__bf16 h) { return (float)h; }

// ---------------------------------------------------------------------------
// TDM: 2-D tensor_load_to_lds. Builds the D# per CDNA5 ISA §8.3/8.4:
//  group0: count=1 | lds_addr[63:32] | global_addr[120:64] | type=2[127:126]
//  group1: wg_mask=0, data_size=1 (2B), tensor_dim0/1, tile_dim0/1,
//          tensor_dim0_stride (elements)
// Issued by one wave; completion via TENSORcnt + workgroup barrier.
// ---------------------------------------------------------------------------
__device__ __forceinline__ void tdm_load_2d_bf16(unsigned lds_addr, const void* gptr,
                                                 unsigned tensor_d0, unsigned tensor_d1,
                                                 unsigned tile_d0, unsigned tile_d1,
                                                 unsigned stride0) {
    unsigned long long ga = (unsigned long long)gptr;
    u32x4 g0;
    g0[0] = 0x1u;                                   // count=1, user mode
    g0[1] = lds_addr;                               // LDS byte address
    g0[2] = (unsigned)ga;                           // global_addr[31:0]
    g0[3] = (unsigned)((ga >> 32) & 0x01FFFFFFu) | (2u << 30);  // ga[56:32] | type=2
    u32x8 g1;
    g1[0] = 0x00010000u;                            // wg_mask=0, data_size=1 (2B)
    g1[1] = (tensor_d0 & 0xFFFFu) << 16;            // abar_addr=0 | tensor_dim0[15:0]
    g1[2] = (tensor_d0 >> 16) | ((tensor_d1 & 0xFFFFu) << 16);
    g1[3] = (tensor_d1 >> 16) | ((tile_d0 & 0xFFFFu) << 16);
    g1[4] = tile_d1 & 0xFFFFu;                      // tile_dim1 | tile_dim2=0
    g1[5] = stride0;                                // tensor_dim0_stride[31:0]
    g1[6] = 0u;                                     // stride0[47:32] | stride1[15:0]
    g1[7] = 0u;
    asm volatile("tensor_load_to_lds %0, %1" :: "s"(g0), "s"(g1) : "memory");
}

// ---------------------------------------------------------------------------
// K0: convert 1x1-conv weights to bf16 once.
// ---------------------------------------------------------------------------
__global__ void __launch_bounds__(256)
cvt_weights_kernel(const float* __restrict__ wqkv, const float* __restrict__ wproj,
                   __bf16* __restrict__ wq_bf, __bf16* __restrict__ wp_bf) {
    int i = blockIdx.x * 256 + threadIdx.x;
    if (i < CH3 * CH) wq_bf[i] = f2bf(wqkv[i]);
    if (i < CH * CH)  wp_bf[i] = f2bf(wproj[i]);
}

// ---------------------------------------------------------------------------
// K1: qkv pointwise conv as WMMA GEMM.
// pre[b,m,p] = sum_k wq[m,k] * x[b,k,p];  M=576, K=192, N=16384 per batch.
// A tile staged by TDM (k-contiguous in global and LDS); B tile staged with
// packed bf16-pair LDS stores.
// ---------------------------------------------------------------------------
__global__ void __launch_bounds__(256)
gemm_qkv_kernel(const __bf16* __restrict__ wq, const float* __restrict__ x,
                __bf16* __restrict__ pre) {
    __shared__ __bf16 lA[64 * 32];    // [m][k], k-contiguous
    __shared__ __bf16 lB[128 * 32];   // [n][k], k-contiguous
    const int t = threadIdx.x;
    const int lane = t & 31, wid = t >> 5;
    const int waveM = wid & 3, waveN = wid >> 2;
    const int p0 = blockIdx.x * 128;
    const int m0 = blockIdx.y * 64;
    const int b  = blockIdx.z;
    const unsigned ldsA = (unsigned)(unsigned long long)(void*)lA;

    FragC acc[4];
    for (int s = 0; s < 4; ++s)
        for (int i = 0; i < 8; ++i) acc[s].f[i] = 0.f;

    for (int kc = 0; kc < 6; ++kc) {
        const int k0 = kc * 32;
        // A tile via Tensor Data Mover: tile 32(k) x 64(m), row stride CH
        if (wid == 0)
            tdm_load_2d_bf16(ldsA, wq + (size_t)m0 * CH + k0,
                             CH, CH3, 32, 64, CH);
        // B tile: pack two consecutive k into one 32-bit LDS store
        for (int i = 0; i < 8; ++i) {
            int idx = t + i * 256;           // 0..2047
            int n  = idx & 127;
            int kp = idx >> 7;               // 0..15
            const float* xp = &x[(((size_t)b * CH + (k0 + 2 * kp)) << 14) + p0 + n];
            PkBf pk;
            pk.h[0] = f2bf(xp[0]);
            pk.h[1] = f2bf(xp[(size_t)1 << 14]);
            *(unsigned*)&lB[n * 32 + 2 * kp] = pk.u;
            if (i == 0 && kc < 5)            // prefetch next chunk's row
                __builtin_prefetch(xp + ((size_t)32 << 14), 0, 1);
        }
        if (wid == 0) __builtin_amdgcn_s_wait_tensorcnt(0);
        __syncthreads();

        FragAB a;
        {
            int m  = waveM * 16 + (lane & 15);
            int ka = (lane < 16) ? 0 : 8;
            a.h[0] = *(const v8bf*)&lA[m * 32 + ka];
            a.h[1] = *(const v8bf*)&lA[m * 32 + ka + 16];
        }
        for (int s = 0; s < 4; ++s) {
            FragAB bb;
            int n  = waveN * 64 + s * 16 + (lane & 15);
            int kb = (lane < 16) ? 0 : 8;
            bb.h[0] = *(const v8bf*)&lB[n * 32 + kb];
            bb.h[1] = *(const v8bf*)&lB[n * 32 + kb + 16];
            acc[s].v = __builtin_amdgcn_wmma_f32_16x16x32_bf16(
                false, a.v, false, bb.v, (short)0, acc[s].v, false, false);
        }
        __syncthreads();
    }
    for (int s = 0; s < 4; ++s) {
        int n  = p0 + waveN * 64 + s * 16 + (lane & 15);
        int mb = m0 + waveM * 16 + ((lane >> 4) & 1) * 8;
        for (int r = 0; r < 8; ++r)
            pre[(((size_t)b * CH3 + mb + r) << 14) + n] = f2bf(acc[s].f[r]);
    }
}

// ---------------------------------------------------------------------------
// K2: depthwise 3x3, SAME padding (memory bound; one thread per output pixel)
// ---------------------------------------------------------------------------
__global__ void __launch_bounds__(256)
dwconv_kernel(const __bf16* __restrict__ pre, const float* __restrict__ wdw,
              __bf16* __restrict__ post) {
    size_t idx = (size_t)blockIdx.x * 256 + threadIdx.x;
    int p = (int)(idx & (HWSZ - 1));
    size_t bc = idx >> 14;
    int ch = (int)(bc % CH3);
    int xx = p & (WIDTH - 1), yy = p >> 7;
    const __bf16* src = pre + (bc << 14);
    float acc = 0.f;
#pragma unroll
    for (int ky = -1; ky <= 1; ++ky) {
        int y2 = yy + ky;
        if ((unsigned)y2 >= WIDTH) continue;
#pragma unroll
        for (int kx = -1; kx <= 1; ++kx) {
            int x2 = xx + kx;
            if ((unsigned)x2 >= WIDTH) continue;
            acc += wdw[ch * 9 + (ky + 1) * 3 + (kx + 1)] * bf2f(src[(y2 << 7) + x2]);
        }
    }
    post[idx] = f2bf(acc);
}

// ---------------------------------------------------------------------------
// K3: inverse L2 norms of q rows (ch 0..191) and k rows (ch 192..383)
// ---------------------------------------------------------------------------
__global__ void __launch_bounds__(256)
norms_kernel(const __bf16* __restrict__ post, float* __restrict__ invq,
             float* __restrict__ invk) {
    __shared__ float red[256];
    int row  = blockIdx.x;                 // 0 .. 2*BATCH*CH-1
    int qsel = (row < BATCH * CH) ? 1 : 0;
    int r2   = qsel ? row : row - BATCH * CH;
    int b = r2 / CH, c = r2 % CH;
    int ch = (qsel ? 0 : CH) + c;
    const __bf16* src = post + (((size_t)b * CH3 + ch) << 14);
    float ss = 0.f;
    for (int i = threadIdx.x; i < HWSZ; i += 256) {
        float v = bf2f(src[i]);
        ss += v * v;
    }
    red[threadIdx.x] = ss;
    __syncthreads();
    for (int s = 128; s > 0; s >>= 1) {
        if ((int)threadIdx.x < s) red[threadIdx.x] += red[threadIdx.x + s];
        __syncthreads();
    }
    if (threadIdx.x == 0) {
        float inv = 1.f / fmaxf(sqrtf(red[0]), 1e-12f);
        (qsel ? invq : invk)[b * CH + c] = inv;
    }
}

// ---------------------------------------------------------------------------
// K4: channel-attention scores. One block per (b,h): 24x24 Gram matrix over
// K=16384 via 2x2 WMMA tiles (padded to 32x32), 4 waves split K, then fold in
// 1/|q| * 1/|k| * temperature and softmax over d.
// ---------------------------------------------------------------------------
__global__ void __launch_bounds__(128)
attn_scores_kernel(const __bf16* __restrict__ postQ, const __bf16* __restrict__ postK,
                   const float* __restrict__ invQ, const float* __restrict__ invK,
                   const float* __restrict__ temperature, float* __restrict__ S) {
    __shared__ float part[4][32][32];
    const int t = threadIdx.x, lane = t & 31, w = t >> 5;
    const int bh = blockIdx.x;
    const int b = bh / HEADS, h = bh % HEADS;
    const __bf16* qbase = postQ + (((size_t)b * CH3 + h * HDIM) << 14);        // q: ch 0..
    const __bf16* kbase = postK + (((size_t)b * CH3 + CH + h * HDIM) << 14);   // k: ch 192..

    FragC acc[2][2];
    for (int i = 0; i < 2; ++i)
        for (int j = 0; j < 2; ++j)
            for (int r = 0; r < 8; ++r) acc[i][j].f[r] = 0.f;

    for (int kc = w; kc < 512; kc += 4) {
        const int k0 = kc * 32;
        FragAB a[2], bb[2];
        for (int tm = 0; tm < 2; ++tm) {
            int m  = tm * 16 + (lane & 15);
            int ka = (lane < 16) ? 0 : 8;
            if (m < HDIM) {
                const __bf16* rp = qbase + ((size_t)m << 14) + k0;
                a[tm].h[0] = *(const v8bf*)(rp + ka);
                a[tm].h[1] = *(const v8bf*)(rp + ka + 16);
            } else {
                for (int e = 0; e < 16; ++e) a[tm].e[e] = f2bf(0.f);
            }
        }
        for (int tn = 0; tn < 2; ++tn) {
            int d  = tn * 16 + (lane & 15);
            int kb = (lane < 16) ? 0 : 8;
            if (d < HDIM) {
                const __bf16* rp = kbase + ((size_t)d << 14) + k0;
                bb[tn].h[0] = *(const v8bf*)(rp + kb);
                bb[tn].h[1] = *(const v8bf*)(rp + kb + 16);
            } else {
                for (int e = 0; e < 16; ++e) bb[tn].e[e] = f2bf(0.f);
            }
        }
        for (int tm = 0; tm < 2; ++tm)
            for (int tn = 0; tn < 2; ++tn)
                acc[tm][tn].v = __builtin_amdgcn_wmma_f32_16x16x32_bf16(
                    false, a[tm].v, false, bb[tn].v, (short)0, acc[tm][tn].v,
                    false, false);
    }
    for (int tm = 0; tm < 2; ++tm)
        for (int tn = 0; tn < 2; ++tn) {
            int col  = tn * 16 + (lane & 15);
            int rowo = tm * 16 + ((lane >> 4) & 1) * 8;
            for (int r = 0; r < 8; ++r)
                part[w][rowo + r][col] = acc[tm][tn].f[r];
        }
    __syncthreads();
    for (int i = t; i < 1024; i += 128) {
        int r = i >> 5, c2 = i & 31;
        part[0][r][c2] += part[1][r][c2] + part[2][r][c2] + part[3][r][c2];
    }
    __syncthreads();
    if (t < HDIM) {
        const int c = t;
        const float iq = invQ[b * CH + h * HDIM + c];
        const float tp = temperature[h];
        float vals[HDIM];
        float mx = -3.4e38f;
#pragma unroll
        for (int d = 0; d < HDIM; ++d) {
            float v = part[0][c][d] * iq * invK[b * CH + h * HDIM + d] * tp;
            vals[d] = v;
            mx = fmaxf(mx, v);
        }
        float sum = 0.f;
#pragma unroll
        for (int d = 0; d < HDIM; ++d) { vals[d] = __expf(vals[d] - mx); sum += vals[d]; }
        float rs = 1.f / sum;
        float* so = S + ((size_t)bh * HDIM + c) * HDIM;
#pragma unroll
        for (int d = 0; d < HDIM; ++d) so[d] = vals[d] * rs;
    }
}

// ---------------------------------------------------------------------------
// K5: out[b,h,c,n] = sum_d S[c,d] * v[b,h,d,n]  (K=24: VALU FMA, LDS broadcast)
// ---------------------------------------------------------------------------
__global__ void __launch_bounds__(256)
attn_apply_kernel(const float* __restrict__ S, const __bf16* __restrict__ postV,
                  __bf16* __restrict__ ao) {
    __shared__ float  lS[HDIM][HDIM];
    __shared__ __bf16 lV[HDIM][256];
    const int t  = threadIdx.x;
    const int n0 = blockIdx.x * 256;
    const int bh = blockIdx.y;
    const int b = bh / HEADS, h = bh % HEADS;
    const float* sp = S + (size_t)bh * HDIM * HDIM;
    for (int i = t; i < HDIM * HDIM; i += 256) ((float*)lS)[i] = sp[i];
    const __bf16* vbase = postV + (((size_t)b * CH3 + 2 * CH + h * HDIM) << 14) + n0;
#pragma unroll
    for (int d = 0; d < HDIM; ++d) lV[d][t] = vbase[((size_t)d << 14) + t];
    __syncthreads();
    __bf16* ob = ao + (((size_t)b * CH + h * HDIM) << 14) + n0 + t;
#pragma unroll 4
    for (int c = 0; c < HDIM; ++c) {
        float accv = 0.f;
#pragma unroll
        for (int d = 0; d < HDIM; ++d) accv += lS[c][d] * bf2f(lV[d][t]);
        ob[(size_t)c << 14] = f2bf(accv);
    }
}

// ---------------------------------------------------------------------------
// K6: projection 1x1 conv as WMMA GEMM, fp32 stores straight to d_out.
// out[b,m,p] = sum_k wp[m,k] * ao[b,k,p]; M=192, K=192, N=16384 per batch.
// ---------------------------------------------------------------------------
__global__ void __launch_bounds__(256)
gemm_proj_kernel(const __bf16* __restrict__ wp, const __bf16* __restrict__ ao,
                 float* __restrict__ outp) {
    __shared__ __bf16 lA[64 * 32];
    __shared__ __bf16 lB[128 * 32];
    const int t = threadIdx.x;
    const int lane = t & 31, wid = t >> 5;
    const int waveM = wid & 3, waveN = wid >> 2;
    const int p0 = blockIdx.x * 128;
    const int m0 = blockIdx.y * 64;
    const int b  = blockIdx.z;
    const unsigned ldsA = (unsigned)(unsigned long long)(void*)lA;

    FragC acc[4];
    for (int s = 0; s < 4; ++s)
        for (int i = 0; i < 8; ++i) acc[s].f[i] = 0.f;

    for (int kc = 0; kc < 6; ++kc) {
        const int k0 = kc * 32;
        if (wid == 0)
            tdm_load_2d_bf16(ldsA, wp + (size_t)m0 * CH + k0,
                             CH, CH, 32, 64, CH);
        for (int i = 0; i < 8; ++i) {
            int idx = t + i * 256;
            int n  = idx & 127;
            int kp = idx >> 7;
            const __bf16* apn = &ao[(((size_t)b * CH + (k0 + 2 * kp)) << 14) + p0 + n];
            PkBf pk;
            pk.h[0] = apn[0];
            pk.h[1] = apn[(size_t)1 << 14];
            *(unsigned*)&lB[n * 32 + 2 * kp] = pk.u;
            if (i == 0 && kc < 5)
                __builtin_prefetch(apn + ((size_t)32 << 14), 0, 1);
        }
        if (wid == 0) __builtin_amdgcn_s_wait_tensorcnt(0);
        __syncthreads();

        FragAB a;
        {
            int m  = waveM * 16 + (lane & 15);
            int ka = (lane < 16) ? 0 : 8;
            a.h[0] = *(const v8bf*)&lA[m * 32 + ka];
            a.h[1] = *(const v8bf*)&lA[m * 32 + ka + 16];
        }
        for (int s = 0; s < 4; ++s) {
            FragAB bb;
            int n  = waveN * 64 + s * 16 + (lane & 15);
            int kb = (lane < 16) ? 0 : 8;
            bb.h[0] = *(const v8bf*)&lB[n * 32 + kb];
            bb.h[1] = *(const v8bf*)&lB[n * 32 + kb + 16];
            acc[s].v = __builtin_amdgcn_wmma_f32_16x16x32_bf16(
                false, a.v, false, bb.v, (short)0, acc[s].v, false, false);
        }
        __syncthreads();
    }
    for (int s = 0; s < 4; ++s) {
        int n  = p0 + waveN * 64 + s * 16 + (lane & 15);
        int mb = m0 + waveM * 16 + ((lane >> 4) & 1) * 8;
        for (int r = 0; r < 8; ++r)
            outp[(((size_t)b * CH + mb + r) << 14) + n] = acc[s].f[r];
    }
}

// ---------------------------------------------------------------------------
extern "C" void kernel_launch(void* const* d_in, const int* in_sizes, int n_in,
                              void* d_out, int out_size, void* d_ws, size_t ws_size,
                              hipStream_t stream) {
    (void)in_sizes; (void)n_in; (void)out_size; (void)ws_size;
    const float* rgb   = (const float*)d_in[0];
    const float* ir    = (const float*)d_in[1];
    const float* wqkv  = (const float*)d_in[2];
    const float* wdw   = (const float*)d_in[3];
    const float* wproj = (const float*)d_in[4];
    const float* temp  = (const float*)d_in[5];

    char* ws = (char*)d_ws;
    size_t off = 0;
    auto alloc = [&](size_t bytes) -> void* {
        void* p = ws + off;
        off = (off + bytes + 255) & ~(size_t)255;
        return p;
    };

    __bf16* wq_bf    = (__bf16*)alloc((size_t)CH3 * CH * 2);
    __bf16* wp_bf    = (__bf16*)alloc((size_t)CH * CH * 2);
    __bf16* pre      = (__bf16*)alloc((size_t)BATCH * CH3 * HWSZ * 2);
    __bf16* post_rgb = (__bf16*)alloc((size_t)BATCH * CH3 * HWSZ * 2);
    __bf16* post_ir  = (__bf16*)alloc((size_t)BATCH * CH3 * HWSZ * 2);
    float*  invq_rgb = (float*)alloc((size_t)BATCH * CH * 4);
    float*  invk_rgb = (float*)alloc((size_t)BATCH * CH * 4);
    float*  invq_ir  = (float*)alloc((size_t)BATCH * CH * 4);
    float*  invk_ir  = (float*)alloc((size_t)BATCH * CH * 4);
    float*  S_ir     = (float*)alloc((size_t)BATCH * HEADS * HDIM * HDIM * 4);
    float*  S_rgb    = (float*)alloc((size_t)BATCH * HEADS * HDIM * HDIM * 4);
    __bf16* ao_ir    = (__bf16*)alloc((size_t)BATCH * CH * HWSZ * 2);
    __bf16* ao_rgb   = (__bf16*)alloc((size_t)BATCH * CH * HWSZ * 2);

    cvt_weights_kernel<<<(CH3 * CH + 255) / 256, 256, 0, stream>>>(wqkv, wproj, wq_bf, wp_bf);

    dim3 gq(HWSZ / 128, CH3 / 64, BATCH);
    unsigned dwblocks = (unsigned)((size_t)BATCH * CH3 * HWSZ / 256);

    gemm_qkv_kernel<<<gq, 256, 0, stream>>>(wq_bf, rgb, pre);
    dwconv_kernel<<<dwblocks, 256, 0, stream>>>(pre, wdw, post_rgb);
    gemm_qkv_kernel<<<gq, 256, 0, stream>>>(wq_bf, ir, pre);
    dwconv_kernel<<<dwblocks, 256, 0, stream>>>(pre, wdw, post_ir);

    norms_kernel<<<2 * BATCH * CH, 256, 0, stream>>>(post_rgb, invq_rgb, invk_rgb);
    norms_kernel<<<2 * BATCH * CH, 256, 0, stream>>>(post_ir, invq_ir, invk_ir);

    attn_scores_kernel<<<BATCH * HEADS, 128, 0, stream>>>(post_rgb, post_ir,
                                                          invq_rgb, invk_ir, temp, S_ir);
    attn_scores_kernel<<<BATCH * HEADS, 128, 0, stream>>>(post_ir, post_rgb,
                                                          invq_ir, invk_rgb, temp, S_rgb);

    dim3 ga(HWSZ / 256, BATCH * HEADS);
    attn_apply_kernel<<<ga, 256, 0, stream>>>(S_ir, post_ir, ao_ir);
    attn_apply_kernel<<<ga, 256, 0, stream>>>(S_rgb, post_rgb, ao_rgb);

    float* out_rgb = (float*)d_out;
    float* out_ir  = out_rgb + (size_t)BATCH * CH * HWSZ;
    dim3 gp(HWSZ / 128, CH / 64, BATCH);
    gemm_proj_kernel<<<gp, 256, 0, stream>>>(wp_bf, ao_rgb, out_rgb);
    gemm_proj_kernel<<<gp, 256, 0, stream>>>(wp_bf, ao_ir, out_ir);
}
